// LiquidTimeConstantCell_50929722196029
// MI455X (gfx1250) — compile-verified, matrix operand order checked by reading, not run on hardware
//
#include <hip/hip_runtime.h>
#include <hip/hip_bf16.h>
#include <math.h>

// ---------------------------------------------------------------------------
// Liquid Time-Constant cell, persistent-state WMMA kernel for gfx1250.
//   B=32768, H=IN=128, T=10.  One wave = 16 batch rows, full 128-wide state.
//   W (recurrent weights) staged to LDS via the Tensor Data Mover.
// ---------------------------------------------------------------------------

typedef __bf16 bf16_t;
typedef __attribute__((ext_vector_type(16))) __bf16 v16bf;
typedef __attribute__((ext_vector_type(8)))  float  v8f;
typedef __attribute__((ext_vector_type(4)))  unsigned int u32x4;
typedef __attribute__((ext_vector_type(8)))  int i32x8;
typedef __attribute__((ext_vector_type(4)))  int i32x4;

#define B_DIM   32768
#define H_DIM   128
#define IN_DIM  128
#define T_DIM   10
#define WAVES   8
#define RPW     16                 // rows per wave
#define RPB     (WAVES * RPW)      // rows per block = 128
#define LSTR    136                // padded LDS row stride (bf16 elems) -> 4-bank skew

// LDS layout (byte offsets into one arena; scratch is aliased by RK4 staging):
//   [0,          34816)  s_weff : W*wiring, bf16, 128 x LSTR
//   [34816,      35328)  s_bias : f32[128]
//   [35328,     100864)  s_scratch : f32 W tile via TDM   (phase 1 only)
//                        s_stage   : 8 waves x 16 x LSTR bf16 (RK4, aliases scratch)
#define OFF_WEFF    0
#define OFF_BIAS    34816
#define OFF_SCRATCH 35328
#define SMEM_BYTES  (OFF_SCRATCH + H_DIM * H_DIM * 4)

__device__ __forceinline__ v8f wmma_bf16(v16bf a, v16bf b, v8f c) {
    // D = A(16x32 bf16) * B(32x16 bf16) + C(16x16 f32)
    return __builtin_amdgcn_wmma_f32_16x16x32_bf16(
        false, a, false, b, (short)0, c, false, false);
}

// Fragment load from LDS bf16 matrix (row-major, padded stride LSTR).
__device__ __forceinline__ v16bf frag_lds(const bf16_t* base, int row0, int k0,
                                          int lane) {
    const int r    = row0 + (lane & 15);
    const int koff = k0 + ((lane >> 4) << 3);
    const bf16_t* p = base + r * LSTR + koff;
    union { v16bf v; uint4 q[2]; } u;
    u.q[0] = *reinterpret_cast<const uint4*>(p);        // K = koff .. koff+7
    u.q[1] = *reinterpret_cast<const uint4*>(p + 16);   // K = koff+16 .. koff+23
    return u.v;
}

// Fragment built from global f32 (row-major, arbitrary stride), converted bf16.
__device__ __forceinline__ v16bf frag_gf32(const float* __restrict__ base,
                                           int row0, int k0, int stride,
                                           int lane) {
    const int r    = row0 + (lane & 15);
    const int koff = k0 + ((lane >> 4) << 3);
    const float* p = base + (size_t)r * stride + koff;
    float4 a = *reinterpret_cast<const float4*>(p);
    float4 b = *reinterpret_cast<const float4*>(p + 4);
    float4 c = *reinterpret_cast<const float4*>(p + 16);
    float4 d = *reinterpret_cast<const float4*>(p + 20);
    v16bf f;
    f[0]=(bf16_t)a.x;  f[1]=(bf16_t)a.y;  f[2]=(bf16_t)a.z;  f[3]=(bf16_t)a.w;
    f[4]=(bf16_t)b.x;  f[5]=(bf16_t)b.y;  f[6]=(bf16_t)b.z;  f[7]=(bf16_t)b.w;
    f[8]=(bf16_t)c.x;  f[9]=(bf16_t)c.y;  f[10]=(bf16_t)c.z; f[11]=(bf16_t)c.w;
    f[12]=(bf16_t)d.x; f[13]=(bf16_t)d.y; f[14]=(bf16_t)d.z; f[15]=(bf16_t)d.w;
    return f;
}

// Store a C-layout (8 tiles x v8f) register block to a row-major [*,128] array.
__device__ __forceinline__ void store_tile(float* __restrict__ dst, int row0,
                                           int lane, const float t[8][8]) {
    const int m0 = row0 + ((lane >> 4) << 3);
    const int nl = lane & 15;
#pragma unroll
    for (int tn = 0; tn < 8; ++tn)
#pragma unroll
        for (int v = 0; v < 8; ++v)
            dst[(size_t)(m0 + v) * H_DIM + tn * 16 + nl] = t[tn][v];
}

__device__ __forceinline__ float sigmoidf_(float z) {
    return 1.0f / (1.0f + __expf(-z));
}

__global__ __launch_bounds__(256)
void ltc_rk4_kernel(const float* __restrict__ x,      const float* __restrict__ hidden,
                    const float* __restrict__ ts,     const float* __restrict__ wiring,
                    const float* __restrict__ W,      const float* __restrict__ b,
                    const float* __restrict__ W_in,   const float* __restrict__ b_in,
                    const float* __restrict__ W_upd,  const float* __restrict__ b_upd,
                    const float* __restrict__ W_rst,  const float* __restrict__ b_rst,
                    float* __restrict__ out,          float* __restrict__ ws_upd)
{
    __shared__ __align__(16) unsigned char s_mem[SMEM_BYTES];
    bf16_t* const s_weff    = (bf16_t*)(s_mem + OFF_WEFF);
    float*  const s_bias    = (float*)(s_mem + OFF_BIAS);
    float*  const s_scratch = (float*)(s_mem + OFF_SCRATCH);   // f32 W (TDM dest)
    bf16_t* const s_stage   = (bf16_t*)(s_mem + OFF_SCRATCH);  // aliases scratch later

    const int tid  = threadIdx.x;
    const int lane = tid & 31;
    const int wave = tid >> 5;
    const int row0 = blockIdx.x * RPB + wave * RPW;
    const int nl   = lane & 15;
    const int m0   = (lane >> 4) << 3;

    // ---- phase 1: DMA raw W into LDS via the Tensor Data Mover -------------
#if __has_builtin(__builtin_amdgcn_tensor_load_to_lds)
    if (wave == 0) {
        const unsigned int lds_addr = (unsigned int)(size_t)(void*)s_scratch;
        const unsigned long long ga = (unsigned long long)(size_t)W;
        // D# group 0: count=1 | lds_addr | global_addr | type=2 ("image")
        u32x4 g0;
        g0[0] = 1u;                                      // count=1, user mode
        g0[1] = lds_addr;                                // LDS byte address
        g0[2] = (unsigned int)(ga & 0xffffffffull);      // global addr [95:64]
        g0[3] = (unsigned int)((ga >> 32) & 0x1ffffffull) | (2u << 30); // + type
        // D# group 1: data_size=4B, 2-D tensor 128x128, tile 128x128, stride 128
        i32x8 g1;
        g1[0] = (int)(2u << 16);                 // data_size=2 (4 bytes), mask=0
        g1[1] = (int)(128u << 16);               // tensor_dim0 low16  @ [63:48]
        g1[2] = (int)(128u << 16);               // tensor_dim1 low16  @ [95:80]
        g1[3] = (int)(128u << 16);               // tile_dim0          @ [127:112]
        g1[4] = (int)128u;                       // tile_dim1          @ [143:128]
        g1[5] = (int)128u;                       // tensor_dim0_stride @ [191:160]
        g1[6] = (int)((16384u & 0xffffu) << 16); // tensor_dim1_stride low16 @ [223:208]
        g1[7] = 0;
        i32x4 z4 = {0, 0, 0, 0};
#if defined(__clang_major__) && (__clang_major__ >= 23)
        i32x8 z8 = {0, 0, 0, 0, 0, 0, 0, 0};
        __builtin_amdgcn_tensor_load_to_lds(g0, g1, z4, z4, z8, 0);
#else
        __builtin_amdgcn_tensor_load_to_lds(g0, g1, z4, z4, 0);
#endif
        __builtin_amdgcn_s_wait_tensorcnt(0);
    }
#else
    for (int i = tid; i < H_DIM * H_DIM; i += blockDim.x) s_scratch[i] = W[i];
#endif
    for (int i = tid; i < H_DIM; i += blockDim.x) s_bias[i] = b[i];
    __syncthreads();

    // ---- W_eff = W * wiring  (bf16, padded rows), from the TDM'd scratch ---
    for (int i = tid; i < H_DIM * H_DIM; i += blockDim.x) {
        int r = i >> 7, c = i & 127;
        s_weff[r * LSTR + c] = (bf16_t)(s_scratch[i] * wiring[i]);
    }
    __syncthreads();   // scratch dead from here on -> s_stage may alias it

    // ---- gates phase: x_proj, update, h0 = reset * hidden ------------------
    v16bf xa[4], ha[4];
#pragma unroll
    for (int kf = 0; kf < 4; ++kf) {
        xa[kf] = frag_gf32(x,      row0, kf * 32, IN_DIM, lane);
        ha[kf] = frag_gf32(hidden, row0, kf * 32, H_DIM,  lane);
    }

    float xp[8][8], up[8][8], h[8][8];
#pragma unroll
    for (int tn = 0; tn < 8; ++tn) {
        const int n = tn * 16 + nl;
        // x_proj = x @ W_in^T + b_in
        v8f acc = {};
#pragma unroll
        for (int kf = 0; kf < 4; ++kf)
            acc = wmma_bf16(xa[kf], frag_gf32(W_in, tn * 16, kf * 32, IN_DIM, lane), acc);
        const float bi = b_in[n];
#pragma unroll
        for (int v = 0; v < 8; ++v) xp[tn][v] = acc[v] + bi;

        // update = sigmoid([hidden, x] @ W_upd^T + b_upd)   (K = 256)
        acc = (v8f){};
#pragma unroll
        for (int kf = 0; kf < 4; ++kf)
            acc = wmma_bf16(ha[kf], frag_gf32(W_upd, tn * 16, kf * 32,       H_DIM + IN_DIM, lane), acc);
#pragma unroll
        for (int kf = 0; kf < 4; ++kf)
            acc = wmma_bf16(xa[kf], frag_gf32(W_upd, tn * 16, 128 + kf * 32, H_DIM + IN_DIM, lane), acc);
        const float bu = b_upd[n];
#pragma unroll
        for (int v = 0; v < 8; ++v) up[tn][v] = sigmoidf_(acc[v] + bu);

        // reset = sigmoid([hidden, x] @ W_rst^T + b_rst);  h0 = reset * hidden
        acc = (v8f){};
#pragma unroll
        for (int kf = 0; kf < 4; ++kf)
            acc = wmma_bf16(ha[kf], frag_gf32(W_rst, tn * 16, kf * 32,       H_DIM + IN_DIM, lane), acc);
#pragma unroll
        for (int kf = 0; kf < 4; ++kf)
            acc = wmma_bf16(xa[kf], frag_gf32(W_rst, tn * 16, 128 + kf * 32, H_DIM + IN_DIM, lane), acc);
        const float br = b_rst[n];
#pragma unroll
        for (int v = 0; v < 8; ++v) {
            float hv = hidden[(size_t)(row0 + m0 + v) * H_DIM + n];
            h[tn][v] = sigmoidf_(acc[v] + br) * hv;      // h0
        }
    }

    // ode_out[0] = h0 (doubles as h0 storage for the final blend),
    // spill update gate to workspace to cap register pressure.
    float* const ode_base = out + (size_t)B_DIM * H_DIM;     // after new_hidden
    store_tile(ode_base, row0, lane, h);
    store_tile(ws_upd,   row0, lane, up);

    // ---- RK4 over 9 intervals ---------------------------------------------
    bf16_t* const st = s_stage + wave * (RPW * LSTR);
    float sum[8][8], arg[8][8], k[8][8];

#pragma unroll 1
    for (int t = 0; t < T_DIM - 1; ++t) {
        const float dt = ts[t + 1] - ts[t];
#pragma unroll
        for (int tn = 0; tn < 8; ++tn)
#pragma unroll
            for (int v = 0; v < 8; ++v) { sum[tn][v] = 0.0f; arg[tn][v] = h[tn][v]; }

#pragma unroll 1
        for (int s = 0; s < 4; ++s) {
            // --- f(arg) = -arg + tanh(arg @ W_eff^T + b) + x_proj ---
            // re-layout arg (C layout) -> A layout via the wave-private LDS tile
#pragma unroll
            for (int tn = 0; tn < 8; ++tn)
#pragma unroll
                for (int v = 0; v < 8; ++v)
                    st[(m0 + v) * LSTR + tn * 16 + nl] = (bf16_t)arg[tn][v];
            asm volatile("" ::: "memory");   // keep DS store->load order (per-wave LDS in-order)

            v16bf af[4];
#pragma unroll
            for (int kf = 0; kf < 4; ++kf) af[kf] = frag_lds(st, 0, kf * 32, lane);

#pragma unroll
            for (int tn = 0; tn < 8; ++tn) {
                v8f acc = {};
#pragma unroll
                for (int kf = 0; kf < 4; ++kf)
                    acc = wmma_bf16(af[kf], frag_lds(s_weff, tn * 16, kf * 32, lane), acc);
                const float bb = s_bias[tn * 16 + nl];
#pragma unroll
                for (int v = 0; v < 8; ++v)
                    k[tn][v] = -arg[tn][v] + tanhf(acc[v] + bb) + xp[tn][v];
            }

            const float wgt = (s == 0 || s == 3) ? 1.0f : 2.0f;
            const float cc  = (s == 2) ? dt : 0.5f * dt;
#pragma unroll
            for (int tn = 0; tn < 8; ++tn)
#pragma unroll
                for (int v = 0; v < 8; ++v) {
                    sum[tn][v] += wgt * k[tn][v];
                    if (s < 3) arg[tn][v] = h[tn][v] + cc * k[tn][v];
                }
        }

        const float f6 = dt * (1.0f / 6.0f);
#pragma unroll
        for (int tn = 0; tn < 8; ++tn)
#pragma unroll
            for (int v = 0; v < 8; ++v) h[tn][v] += f6 * sum[tn][v];

        // ode_out[t+1] = h
        store_tile(ode_base + (size_t)(t + 1) * B_DIM * H_DIM, row0, lane, h);
    }

    // ---- new_hidden = update * h_last + (1 - update) * h0 ------------------
#pragma unroll
    for (int tn = 0; tn < 8; ++tn) {
        const int n = tn * 16 + nl;
#pragma unroll
        for (int v = 0; v < 8; ++v) {
            const size_t idx = (size_t)(row0 + m0 + v) * H_DIM + n;
            const float u   = ws_upd[idx];
            const float h0v = ode_base[idx];                 // ode_out[0] == h0
            out[idx] = u * h[tn][v] + (1.0f - u) * h0v;
        }
    }
}

extern "C" void kernel_launch(void* const* d_in, const int* in_sizes, int n_in,
                              void* d_out, int out_size, void* d_ws, size_t ws_size,
                              hipStream_t stream) {
    const float* x      = (const float*)d_in[0];
    const float* hidden = (const float*)d_in[1];
    const float* ts     = (const float*)d_in[2];
    const float* wiring = (const float*)d_in[3];
    const float* W      = (const float*)d_in[4];
    const float* b      = (const float*)d_in[5];
    const float* W_in   = (const float*)d_in[6];
    const float* b_in   = (const float*)d_in[7];
    const float* W_upd  = (const float*)d_in[8];
    const float* b_upd  = (const float*)d_in[9];
    const float* W_rst  = (const float*)d_in[10];
    const float* b_rst  = (const float*)d_in[11];

    float* out    = (float*)d_out;
    float* ws_upd = (float*)d_ws;     // B*H f32 scratch for the update gate

    dim3 grid(B_DIM / RPB);           // 256 blocks
    dim3 block(256);                  // 8 waves (wave32)
    hipLaunchKernelGGL(ltc_rk4_kernel, grid, block, 0, stream,
                       x, hidden, ts, wiring, W, b, W_in, b_in,
                       W_upd, b_upd, W_rst, b_rst, out, ws_upd);
}
